// NDDE_54279796687218
// MI455X (gfx1250) — matrix operand order repaired and checked by reading, not compiled, not run on hardware
//
#include <hip/hip_runtime.h>
#include <hip/hip_bf16.h>
#include <math.h>

// ---------------------------------------------------------------------------
// NDDE forward-Euler solve on gfx1250 (MI455X), bf16 WMMA + fp32 accumulate.
// B=1024 rows are independent -> 64 persistent workgroups x 16 rows, each
// runs all 500 steps with only intra-WG barriers. Delay line = LDS ring.
// fp32 state lives in per-lane registers (same lane owns the GEMM2 output
// element and the Euler update for it). Native v_tanh_f32 when available.
// ---------------------------------------------------------------------------

typedef __attribute__((ext_vector_type(16))) __bf16 v16bf;
typedef __attribute__((ext_vector_type(8)))  __bf16 v8bf;
typedef __attribute__((ext_vector_type(8)))  float  v8f;

#define B_      1024
#define D_      128
#define H_      512
#define NSTEPS  500
#define NTAU    10

#if __has_builtin(__builtin_amdgcn_tanhf)
__device__ __forceinline__ float fast_tanh(float x) { return __builtin_amdgcn_tanhf(x); }
#elif __has_builtin(__builtin_amdgcn_tanh_f32)
__device__ __forceinline__ float fast_tanh(float x) { return __builtin_amdgcn_tanh_f32(x); }
#else
__device__ __forceinline__ float fast_tanh(float x) { return tanhf(x); }
#endif

// workspace layout (bytes):
//   [0,       262144)  W1bf [512][256] bf16   (x-part cols 0..127, y-part 128..255)
//   [262144,  393216)  W2bf [128][512] bf16
//   [393216,  395264)  w1t  [512]      f32    (t-column of W1, kept fp32)

__global__ void ndde_prep(const float* __restrict__ W1, const float* __restrict__ W2,
                          __bf16* __restrict__ W1bf, __bf16* __restrict__ W2bf,
                          float* __restrict__ w1t) {
  int i = blockIdx.x * blockDim.x + threadIdx.x;
  if (i < H_ * 256) {                       // W1 row-major [512][257] -> [512][256] bf16
    int n = i >> 8, k = i & 255;
    W1bf[i] = (__bf16)W1[n * 257 + k];
  }
  if (i < H_) w1t[i] = W1[i * 257 + 256];   // t column
  if (i < D_ * H_) W2bf[i] = (__bf16)W2[i]; // W2 row-major [128][512]
}

// A fragment (16x32, 16-bit): lane l holds row M=l%16; lanes 0-15 carry
// K = kb+[0..7] and kb+16+[0..7]; lanes 16-31 carry kb+[8..15] and kb+24+[..7].
// Two contiguous 16-byte chunks per lane.
__device__ __forceinline__ v16bf cat8(v8bf a, v8bf b) {
  return __builtin_shufflevector(a, b, 0,1,2,3,4,5,6,7,8,9,10,11,12,13,14,15);
}
__device__ __forceinline__ v16bf load_a_frag(const __bf16* base, int stride,
                                             int kbase, int m16, int kh) {
  const __bf16* p = base + m16 * stride + kbase + kh * 8;
  return cat8(*(const v8bf*)p, *(const v8bf*)(p + 16));
}
// B fragment (32x16, 16-bit): lane l holds column N=l%16; lanes 0-15 carry
// K = kb+[0..7] and kb+16+[0..7]; lanes 16-31 carry kb+[8..15] and kb+24+[..7].
// B[k][n] = W[row=n][k], so each lane reads two 16-byte chunks of a weight row.
__device__ __forceinline__ v16bf load_b_frag(const __bf16* base, int stride,
                                             int row, int kbase, int kh) {
  const __bf16* p = base + row * stride + kbase + kh * 8;
  return cat8(*(const v8bf*)p, *(const v8bf*)(p + 16));
}
__device__ __forceinline__ v8f wmma_bf16(v16bf a, v16bf b, v8f c) {
  return __builtin_amdgcn_wmma_f32_16x16x32_bf16(false, a, false, b,
                                                 (short)0, c, false, false);
}

__global__ void __launch_bounds__(256)
ndde_main(const float* __restrict__ x0, const float* __restrict__ tau,
          const float* __restrict__ b1, const float* __restrict__ b2,
          const __bf16* __restrict__ W1bf, const __bf16* __restrict__ W2bf,
          const float* __restrict__ w1t, float* __restrict__ out) {
  // padded strides: (stride*2B) % 256B != 0 so strided fragment loads rotate banks
  __shared__ __bf16 s_ring[NTAU + 1][16][136];  // bf16 delay states  (47.9 KB)
  __shared__ __bf16 s_h[16][520];               // bf16 hidden tile   (16.6 KB)

  const int tid  = threadIdx.x;
  const int w    = tid >> 5;        // wave id 0..7 (wave32)
  const int lane = tid & 31;
  const int m16  = lane & 15;
  const int kh   = lane >> 4;
  const int rowbase = blockIdx.x * 16;

  // ---- init: fill all ring slots with x0 (constant history), emit step 0
  for (int i = tid; i < 16 * D_; i += 256) {
    int m = i >> 7, c = i & 127;
    float v = x0[(size_t)(rowbase + m) * D_ + c];
    __bf16 bv = (__bf16)v;
#pragma unroll
    for (int s = 0; s <= NTAU; ++s) s_ring[s][m][c] = bv;
    out[(size_t)(rowbase + m) * D_ + c] = v;
  }

  // per-wave invariants: GEMM1 covers 4 of 32 N-tiles, GEMM2 exactly 1 of 8
  int nb1[4]; float b1v[4], w1tv[4];
#pragma unroll
  for (int i = 0; i < 4; ++i) {
    nb1[i]  = (w + 8 * i) * 16;
    b1v[i]  = b1[nb1[i] + m16];
    w1tv[i] = w1t[nb1[i] + m16];
  }
  const int   nb2 = w * 16;
  const int   col = nb2 + m16;      // the x column this lane owns
  const float b2v = b2[col];
  const float dt  = tau[0] * (1.0f / (float)NTAU);

  // fp32 state in registers: element (m = v+8*kh, col) matches GEMM2 C layout
  float xreg[8];
#pragma unroll
  for (int v = 0; v < 8; ++v)
    xreg[v] = x0[(size_t)(rowbase + v + 8 * kh) * D_ + col];

  const v8f vzero = {};
  __syncthreads();

  for (int n = 0; n < NSTEPS; ++n) {
    const int cur = n % (NTAU + 1);         // slot of x_n
    const int del = (n + 1) % (NTAU + 1);   // slot of x_{n-10}; reused for x_{n+1}
    const float t = (float)n * dt;

    // ---- GEMM1: h_pre = x@W1x^T + y@W1y^T   (K = 8 slabs of 32)
    v8f acc[4] = {vzero, vzero, vzero, vzero};
#pragma unroll
    for (int s = 0; s < 8; ++s) {
      const __bf16* src = (s < 4) ? &s_ring[cur][0][0] : &s_ring[del][0][0];
      v16bf A = load_a_frag(src, 136, (s & 3) * 32, m16, kh);
      const int kw = s * 32;                // W1bf column base (x cols 0..127, y 128..255)
#pragma unroll
      for (int i = 0; i < 4; ++i) {
        v16bf Bf = load_b_frag(W1bf, 256, nb1[i] + m16, kw, kh);
        acc[i] = wmma_bf16(A, Bf, acc[i]);
      }
    }
    // ---- bias (with folded t-column) + native tanh, store bf16 h tile to LDS
#pragma unroll
    for (int i = 0; i < 4; ++i) {
      const float be = b1v[i] + t * w1tv[i];
#pragma unroll
      for (int v = 0; v < 8; ++v) {
        const int m = v + 8 * kh;           // C layout: M = vgpr + 8*(lane/16)
        s_h[m][nb1[i] + m16] = (__bf16)fast_tanh(acc[i][v] + be);
      }
    }
    __syncthreads();                        // h complete; all ring reads done

    // ---- GEMM2: out = h @ W2^T   (K = 16 slabs of 32; W2 frags read from global W2bf each step)
    v8f acc2 = vzero;
#pragma unroll
    for (int s = 0; s < 16; ++s) {
      v16bf A  = load_a_frag(&s_h[0][0], 520, s * 32, m16, kh);
      v16bf Bf = load_b_frag(W2bf, 512, col, s * 32, kh);
      acc2 = wmma_bf16(A, Bf, acc2);
    }
    // ---- fp32 Euler update in registers, refresh ring slot, emit trajectory
#pragma unroll
    for (int v = 0; v < 8; ++v) {
      const int m = v + 8 * kh;
      xreg[v] += dt * (acc2[v] + b2v);
      s_ring[del][m][col] = (__bf16)xreg[v];
      out[(size_t)(n + 1) * (size_t)(B_ * D_) + (size_t)(rowbase + m) * D_ + col] = xreg[v];
    }
    __syncthreads();                        // ring updated before next step reads
  }
}

extern "C" void kernel_launch(void* const* d_in, const int* in_sizes, int n_in,
                              void* d_out, int out_size, void* d_ws, size_t ws_size,
                              hipStream_t stream) {
  const float* x0  = (const float*)d_in[0];
  const float* tau = (const float*)d_in[1];
  const float* W1  = (const float*)d_in[2];
  const float* b1  = (const float*)d_in[3];
  const float* W2  = (const float*)d_in[4];
  const float* b2  = (const float*)d_in[5];
  float* out = (float*)d_out;

  char* ws = (char*)d_ws;
  __bf16* W1bf = (__bf16*)(ws);
  __bf16* W2bf = (__bf16*)(ws + 262144);
  float*  w1t  = (float*)(ws + 393216);

  ndde_prep<<<512, 256, 0, stream>>>(W1, W2, W1bf, W2bf, w1t);      // 131072 threads
  ndde_main<<<B_ / 16, 256, 0, stream>>>(x0, tau, b1, b2, W1bf, W2bf, w1t, out);
}
